// FlowingContext_62715112456629
// MI455X (gfx1250) — matrix operand order repaired and checked by reading, not compiled, not run on hardware
//
#include <hip/hip_runtime.h>
#include <stdint.h>

typedef __bf16 bf16;
typedef __attribute__((ext_vector_type(16))) __bf16 v16bf;
typedef __attribute__((ext_vector_type(8)))  float  v8f;
typedef uint32_t u32x4 __attribute__((ext_vector_type(4)));
typedef uint32_t u32x8 __attribute__((ext_vector_type(8)));

#define SEQ   2048
#define BATCH 8
#define HDIM  1024
#define NTOT  (BATCH * SEQ)          // 16384

// ---------- helpers ----------
__device__ __forceinline__ bf16 f2bf(float f) {
    union { float f; uint32_t u; } v; v.f = f;
    uint32_t r = v.u + 0x7FFFu + ((v.u >> 16) & 1u);   // RNE
    uint16_t h = (uint16_t)(r >> 16);
    return __builtin_bit_cast(bf16, h);
}
__device__ __forceinline__ float bf2f(bf16 h) {
    uint32_t u = ((uint32_t)__builtin_bit_cast(uint16_t, h)) << 16;
    return __builtin_bit_cast(float, u);
}
__device__ __forceinline__ float gelu_exact(float x) {
    return 0.5f * x * (1.0f + erff(x * 0.7071067811865475f));
}
// low 32 bits of a flat shared pointer == LDS (group-relative) byte offset
__device__ __forceinline__ uint32_t lds_off(const void* p) {
    return (uint32_t)(uintptr_t)p;
}
// async global->LDS copy, 16B per lane, tracked on ASYNCcnt
__device__ __forceinline__ void async_b128(uint32_t lds, const bf16* g) {
    asm volatile("global_load_async_to_lds_b128 %0, %1, off"
                 :: "v"(lds), "v"(g) : "memory");
}
// Tensor Data Mover: 2D tile load (2-group D#), tracked on TENSORcnt
__device__ __forceinline__ void tdm_load_to_lds(uint32_t lds, const void* gptr, u32x8 g1) {
    uint64_t ga = (uint64_t)(uintptr_t)gptr;
    u32x4 g0;
    g0[0] = 1u;                                   // count = 1 (valid descriptor)
    g0[1] = lds;                                  // lds_addr
    g0[2] = (uint32_t)ga;                         // global_addr[31:0]
    g0[3] = (uint32_t)(ga >> 32) | 0x80000000u;   // global_addr[56:32] | type=2
    asm volatile("tensor_load_to_lds %0, %1" :: "s"(g0), "s"(g1) : "memory");
}

// ---------- prep kernels ----------
__global__ void cvt_f32_to_bf16_x4(const float* __restrict__ in,
                                   bf16* __restrict__ out, int n4) {
    int i = blockIdx.x * blockDim.x + threadIdx.x;
    if (i >= n4) return;
    float4 v = reinterpret_cast<const float4*>(in)[i];
    union { uint16_t u[4]; uint2 q; } o;
    o.u[0] = __builtin_bit_cast(uint16_t, f2bf(v.x));
    o.u[1] = __builtin_bit_cast(uint16_t, f2bf(v.y));
    o.u[2] = __builtin_bit_cast(uint16_t, f2bf(v.z));
    o.u[3] = __builtin_bit_cast(uint16_t, f2bf(v.w));
    reinterpret_cast<uint2*>(out)[i] = o.q;
}

// w[m][c][t] (fp32) -> Wp[m][t*C + c] (bf16)
template<int CIN, int TAPS>
__global__ void pack_weights(const float* __restrict__ w, bf16* __restrict__ wp, int total) {
    int o = blockIdx.x * blockDim.x + threadIdx.x;
    if (o >= total) return;
    int m = o / (CIN * TAPS);
    int r = o - m * (CIN * TAPS);
    int t = r / CIN;
    int c = r - t * CIN;
    wp[o] = f2bf(w[(size_t)m * CIN * TAPS + (size_t)c * TAPS + t]);
}

// ---------- WMMA conv-as-GEMM with TDM (A) + async-to-LDS (B) staging ----------
// X: [NTOT][CIN] bf16 (channel contiguous).  Wp: [MOUT][TAPS*CIN] bf16.
// Y[n][m] = gelu( bias[m] + sum_{t,c} Wp[m][t*CIN+c] * X[n + t - PAD][c] )
// Block = 4 waves -> 64(M) x 64(N) tile; each wave one 16(M) x 64(N) slice.
template<int CIN, int MOUT, int TAPS, int PAD>
__global__ __launch_bounds__(128)
void conv_gemm_wmma(const bf16* __restrict__ X,
                    const bf16* __restrict__ Wp,
                    const float* __restrict__ bias,
                    bf16* __restrict__ Y) {
    __shared__ bf16 Bs[2][64 * 32];   // [buf][col*32 + c]  4KB each
    __shared__ bf16 As[2][64 * 32];   // [buf][row*32 + c]  4KB each

    const int tid  = threadIdx.y * 32 + threadIdx.x;
    const int lane = threadIdx.x;
    const int lm   = lane & 15;
    const int lh   = lane >> 4;
    const int wave = threadIdx.y;
    const int mblk = blockIdx.x * 64;
    const int m0   = mblk + wave * 16;
    const int n0   = blockIdx.y * 64;
    const int Ktot = TAPS * CIN;
    constexpr int ITERS = CIN / 32;

    // B-copy role: column cb (0..63), 32B half hb
    const int cb = tid >> 1, hb = tid & 1;
    const int ncol = n0 + cb;
    const int bcol = ncol >> 11, scol = ncol & (SEQ - 1);
    const uint32_t bs0 = lds_off(&Bs[0][cb * 32 + hb * 16]);
    const uint32_t bs1 = lds_off(&Bs[1][cb * 32 + hb * 16]);
    const uint32_t as0 = lds_off(&As[0][0]);
    const uint32_t as1 = lds_off(&As[1][0]);

    // TDM group-1 descriptor: data_size=2B, tile 32(K) x 64(M rows), row stride Ktot
    u32x8 g1;
    g1[0] = 0x00010000u;                                        // data_size = 2 bytes
    g1[1] = ((uint32_t)Ktot & 0xFFFFu) << 16;                   // tensor_dim0[15:0]
    g1[2] = ((uint32_t)Ktot >> 16) | (((uint32_t)MOUT & 0xFFFFu) << 16);
    g1[3] = ((uint32_t)MOUT >> 16) | (32u << 16);               // tile_dim0 = 32
    g1[4] = 64u;                                                // tile_dim1 = 64
    g1[5] = (uint32_t)Ktot;                                     // dim0 stride lo32
    g1[6] = 0u;
    g1[7] = 0u;

    v8f acc[4] = {v8f{}, v8f{}, v8f{}, v8f{}};

    int nn[4];
#pragma unroll
    for (int u = 0; u < 4; ++u) nn[u] = n0 + 16 * u + lm;

    for (int t = 0; t < TAPS; ++t) {
        const int  sp    = scol + t - PAD;
        const bool valid = ((unsigned)sp < (unsigned)SEQ);
        const bf16* gsrc = X + (size_t)((bcol << 11) + sp) * CIN + hb * 16;
        const bf16* wsrc = Wp + (size_t)mblk * Ktot + t * CIN;

        // zero padded columns (validity constant across the K loop of this tap)
        if (!valid) {
            uint4 z{0u, 0u, 0u, 0u};
            *reinterpret_cast<uint4*>(&Bs[0][cb * 32 + hb * 16])     = z;
            *reinterpret_cast<uint4*>(&Bs[0][cb * 32 + hb * 16 + 8]) = z;
            *reinterpret_cast<uint4*>(&Bs[1][cb * 32 + hb * 16])     = z;
            *reinterpret_cast<uint4*>(&Bs[1][cb * 32 + hb * 16 + 8]) = z;
        }
        __syncthreads();

        // prologue: stage chunk 0
        if (valid) { async_b128(bs0, gsrc); async_b128(bs0 + 16, gsrc + 8); }
        if (wave == 0) tdm_load_to_lds(as0, wsrc, g1);

        for (int i = 0; i < ITERS; ++i) {
            const int c1 = (i + 1) * 32;
            if (i + 1 < ITERS) {
                const uint32_t bsn = ((i + 1) & 1) ? bs1 : bs0;
                if (valid) { async_b128(bsn, gsrc + c1); async_b128(bsn + 16, gsrc + c1 + 8); }
                if (wave == 0) tdm_load_to_lds(((i + 1) & 1) ? as1 : as0, wsrc + c1, g1);
                asm volatile("s_wait_asynccnt 0x2" ::: "memory");
                if (wave == 0) __builtin_amdgcn_s_wait_tensorcnt((short)1);
            } else {
                asm volatile("s_wait_asynccnt 0x0" ::: "memory");
                if (wave == 0) __builtin_amdgcn_s_wait_tensorcnt((short)0);
            }
            __syncthreads();

            // A fragment: row (wave*16+lm), K = {8lh..8lh+7} U {16+8lh..16+8lh+7}
            const bf16* ap = &As[i & 1][(wave * 16 + lm) * 32];
            union { v16bf v; uint4 q[2]; } A;
            A.q[0] = *reinterpret_cast<const uint4*>(ap + 8 * lh);
            A.q[1] = *reinterpret_cast<const uint4*>(ap + 16 + 8 * lh);
#pragma unroll
            for (int u = 0; u < 4; ++u) {
                // B fragment: col (u*16+lm), K = {8lh..8lh+7} U {16+8lh..16+8lh+7}
                const bf16* bp = &Bs[i & 1][(u * 16 + lm) * 32];
                union { v16bf v; uint4 q[2]; } Bf;
                Bf.q[0] = *reinterpret_cast<const uint4*>(bp + 8 * lh);
                Bf.q[1] = *reinterpret_cast<const uint4*>(bp + 16 + 8 * lh);
                acc[u] = __builtin_amdgcn_wmma_f32_16x16x32_bf16(
                    false, A.v, false, Bf.v, (short)0, acc[u], false, false);
            }
            __syncthreads();   // all waves done reading buf[i&1] before refill
        }
    }

    // D layout: lane holds col N = lm, rows M = 8*lh + r
#pragma unroll
    for (int u = 0; u < 4; ++u) {
        bf16* orow = Y + (size_t)nn[u] * MOUT + m0 + 8 * lh;
        union { uint4 q; uint16_t h[8]; } o;
#pragma unroll
        for (int r = 0; r < 8; ++r) {
            float v = acc[u][r] + bias[m0 + 8 * lh + r];
            o.h[r] = __builtin_bit_cast(uint16_t, f2bf(gelu_exact(v)));
        }
        *reinterpret_cast<uint4*>(orow) = o.q;
    }
}

// ---------- conv3 + sigmoid + soft mask ----------
__global__ __launch_bounds__(128)
void conv3_relevance(const bf16* __restrict__ Y2,      // [NTOT][512]
                     const float* __restrict__ w3,     // [512][3]
                     const float* __restrict__ b3,
                     const int* __restrict__ mask,
                     float* __restrict__ soft_mask_out,
                     float* __restrict__ scores) {
    __shared__ float red[128];
    const int n = blockIdx.x;
    const int b = n >> 11, s = n & (SEQ - 1);
    float p = 0.f;
#pragma unroll
    for (int t = 0; t < 3; ++t) {
        int sp = s + t - 1;
        if ((unsigned)sp < (unsigned)SEQ) {
            const bf16* row = Y2 + (size_t)(b * SEQ + sp) * 512;
            for (int c = threadIdx.x; c < 512; c += 128)
                p += w3[c * 3 + t] * bf2f(row[c]);
        }
    }
    red[threadIdx.x] = p;
    __syncthreads();
    for (int off = 64; off > 0; off >>= 1) {
        if (threadIdx.x < off) red[threadIdx.x] += red[threadIdx.x + off];
        __syncthreads();
    }
    if (threadIdx.x == 0) {
        float rel = red[0] + b3[0];
        bool m = (mask[n] != 0);
        if (!m) rel = -1e9f;
        float r = 1.f / (1.f + expf(-rel));                          // relevance
        soft_mask_out[n] = 1.f / (1.f + expf(-(r - 0.5f) / 0.65f));  // soft_mask
        scores[n] = m ? r : -__builtin_inff();
    }
}

// ---------- greedy NMS top-4, one block per batch ----------
__global__ __launch_bounds__(256)
void nms_topk(const float* __restrict__ scores,
              float* __restrict__ seg_out_f,
              int* __restrict__ seg_out_i) {
    __shared__ float sc[SEQ];
    __shared__ float rv[256];
    __shared__ int   ri[256];
    __shared__ int   cur;
    const int b = blockIdx.x;
    const int tid = threadIdx.x;
    for (int i = tid; i < SEQ; i += 256) sc[i] = scores[b * SEQ + i];
    __syncthreads();
    for (int k = 0; k < 4; ++k) {
        float bv = -__builtin_inff(); int bi = 0;
        for (int i = tid; i < SEQ; i += 256) {
            float v = sc[i];
            if (v > bv || (v == bv && i < bi)) { bv = v; bi = i; }
        }
        rv[tid] = bv; ri[tid] = bi;
        __syncthreads();
        for (int off = 128; off > 0; off >>= 1) {
            if (tid < off) {
                float v2 = rv[tid + off]; int i2 = ri[tid + off];
                if (v2 > rv[tid] || (v2 == rv[tid] && i2 < ri[tid])) {
                    rv[tid] = v2; ri[tid] = i2;
                }
            }
            __syncthreads();
        }
        if (tid == 0) {
            cur = ri[0];
            seg_out_f[b * 4 + k] = (float)ri[0];
            seg_out_i[b * 4 + k] = ri[0];
        }
        __syncthreads();
        int idx = cur;
        for (int i = tid; i < SEQ; i += 256) {
            int d = (i > idx) ? (i - idx) : (idx - i);
            if (d <= 16) sc[i] = -__builtin_inff();
        }
        __syncthreads();
    }
}

// ---------- combined = lambda * 10 * sm^2 * clip(sum exp(-|d|/8),0,1) ----------
__global__ void combined_kernel(const float* __restrict__ sm,
                                const int* __restrict__ segs,
                                const float* __restrict__ lam,
                                float* __restrict__ comb) {
    int n = blockIdx.x * 256 + threadIdx.x;
    if (n >= NTOT) return;
    int b = n >> 11, s = n & (SEQ - 1);
    float acc = 0.f;
#pragma unroll
    for (int k = 0; k < 4; ++k) {
        float d = fabsf((float)(s - segs[b * 4 + k]));
        acc += expf(-d * 0.125f);
    }
    acc = fminf(fmaxf(acc, 0.f), 1.f);
    float m = sm[n];
    comb[n] = lam[0] * 10.f * m * m * acc;
}

// ---------- bias broadcast: bias[b,0,i,:] = comb[b,:]  (134 MB, pure BW) ----------
__global__ __launch_bounds__(256)
void bias_broadcast(const float* __restrict__ comb, float* __restrict__ bias) {
    int bi = blockIdx.x;                 // b*SEQ + i
    int b = bi >> 11;
    const float4* src = reinterpret_cast<const float4*>(comb + (size_t)b * SEQ);
    float4* dst = reinterpret_cast<float4*>(bias + (size_t)bi * SEQ);
    int t = threadIdx.x;
    dst[t]       = src[t];
    dst[t + 256] = src[t + 256];
}

// ---------- launch ----------
extern "C" void kernel_launch(void* const* d_in, const int* in_sizes, int n_in,
                              void* d_out, int out_size, void* d_ws, size_t ws_size,
                              hipStream_t stream) {
    const float* hidden = (const float*)d_in[0];
    const int*   mask   = (const int*)d_in[1];
    const float* w1     = (const float*)d_in[2];
    const float* b1     = (const float*)d_in[3];
    const float* w2     = (const float*)d_in[4];
    const float* b2     = (const float*)d_in[5];
    const float* w3     = (const float*)d_in[6];
    const float* b3     = (const float*)d_in[7];
    const float* lam    = (const float*)d_in[8];

    char* ws = (char*)d_ws;
    size_t off = 0;
    auto carve = [&](size_t bytes) {
        char* p = ws + off;
        off = (off + bytes + 255) & ~(size_t)255;
        return p;
    };
    bf16* xbf    = (bf16*)carve((size_t)NTOT * HDIM * 2);       // 33.5 MB
    bf16* wp1    = (bf16*)carve((size_t)1024 * 3072 * 2);       //  6.3 MB
    bf16* wp2    = (bf16*)carve((size_t)512 * 5120 * 2);        //  5.2 MB
    bf16* y1     = (bf16*)carve((size_t)NTOT * 1024 * 2);       // 33.5 MB
    bf16* y2     = (bf16*)carve((size_t)NTOT * 512 * 2);        // 16.8 MB
    float* scores= (float*)carve((size_t)NTOT * 4);
    int*  segi   = (int*)carve(32 * 4);
    float* comb  = (float*)carve((size_t)NTOT * 4);

    float* sm_out   = (float*)d_out;                 // [B,S]  16384
    float* seg_out  = (float*)d_out + NTOT;          // [B,4]  32
    float* bias_out = (float*)d_out + NTOT + 32;     // [B,1,S,S]

    // 1) precision conversion + weight packing
    {
        int n4 = (NTOT * HDIM) / 4;
        cvt_f32_to_bf16_x4<<<dim3((n4 + 255) / 256), dim3(256), 0, stream>>>(hidden, xbf, n4);
    }
    pack_weights<1024, 3><<<dim3((1024 * 3072 + 255) / 256), dim3(256), 0, stream>>>(w1, wp1, 1024 * 3072);
    pack_weights<1024, 5><<<dim3((512 * 5120 + 255) / 256), dim3(256), 0, stream>>>(w2, wp2, 512 * 5120);

    // 2) conv1: 1024ch k3 -> gelu   (M=1024, K=3072, N=16384)
    conv_gemm_wmma<1024, 1024, 3, 1>
        <<<dim3(1024 / 64, NTOT / 64), dim3(32, 4), 0, stream>>>(xbf, wp1, b1, y1);

    // 3) conv2: 512ch k5 -> gelu    (M=512, K=5120, N=16384)
    conv_gemm_wmma<1024, 512, 5, 2>
        <<<dim3(512 / 64, NTOT / 64), dim3(32, 4), 0, stream>>>(y1, wp2, b2, y2);

    // 4) conv3 + sigmoid + soft mask
    conv3_relevance<<<dim3(NTOT), dim3(128), 0, stream>>>(y2, w3, b3, mask, sm_out, scores);

    // 5) greedy NMS top-4
    nms_topk<<<dim3(BATCH), dim3(256), 0, stream>>>(scores, seg_out, segi);

    // 6) combined weights
    combined_kernel<<<dim3(NTOT / 256), dim3(256), 0, stream>>>(sm_out, segi, lam, comb);

    // 7) broadcast bias (134 MB)
    bias_broadcast<<<dim3(NTOT), dim3(256), 0, stream>>>(comb, bias_out);
}
